// S5SSM_558345748806
// MI455X (gfx1250) — matrix operand order repaired and checked by reading, not compiled, not run on hardware
//
#include <hip/hip_runtime.h>

typedef __attribute__((ext_vector_type(16))) __bf16 v16bf;
typedef __attribute__((ext_vector_type(8)))  float  v8f;

#define L_SEQ 32768
#define H_DIM 512
#define P_DIM 256
#define N2P   512          // 2*P (re|im concatenated)

#define BM 128
#define BN 256
#define BK 32
#define LDT_S 40           // LDS tile stride (40 elems = 80B: 16B-aligned, padded)

#define CHUNK 256
#define NCH   (L_SEQ / CHUNK)   // 128

static __device__ inline __bf16 f2bf(float f) { return (__bf16)f; }

union FragU { v16bf v; uint4 q[2]; };

// ---------------------------------------------------------------------------
// Packing / casting kernels.
// B operands stored TRANSPOSED ([N][K]) so LDS tiles & fragments match A-path.
//   W1t [2P][H]: rows 0..P-1 = B_re, rows P..2P-1 = B_im   (verbatim copy!)
//   W2t [H][2P]: W2t[h][p] = 2*C_re[h,p], W2t[h][P+p] = -2*C_im[h,p]
// ---------------------------------------------------------------------------
__global__ __launch_bounds__(256) void prep_small(
    const float* __restrict__ log_step,
    const float* __restrict__ B_re, const float* __restrict__ B_im,
    const float* __restrict__ C_re, const float* __restrict__ C_im,
    float* __restrict__ stepb, __bf16* __restrict__ W1t, __bf16* __restrict__ W2t)
{
  int idx = blockIdx.x * 256 + threadIdx.x;     // 0 .. P*H-1
  if (idx < P_DIM) stepb[idx] = __expf(log_step[idx]);
  // W1t: verbatim bf16 copy of B_re then B_im (both [P][H] row-major)
  W1t[idx]                         = f2bf(B_re[idx]);
  W1t[(size_t)P_DIM * H_DIM + idx] = f2bf(B_im[idx]);
  // W2t: [H][2P] from C_re/C_im [H][P] row-major (coalesced in p)
  int h = idx >> 8;              // 0 .. H-1
  int p = idx & (P_DIM - 1);
  W2t[h * N2P + p]         = f2bf( 2.0f * C_re[idx]);
  W2t[h * N2P + P_DIM + p] = f2bf(-2.0f * C_im[idx]);
}

__global__ __launch_bounds__(256) void cast_u(const float* __restrict__ u,
                                              __bf16* __restrict__ ubf)
{
  int i = blockIdx.x * 256 + threadIdx.x;       // one float4 per thread
  float4 v = reinterpret_cast<const float4*>(u)[i];
  __bf16* d = ubf + (size_t)i * 4;
  d[0] = f2bf(v.x); d[1] = f2bf(v.y); d[2] = f2bf(v.z); d[3] = f2bf(v.w);
}

// ---------------------------------------------------------------------------
// WMMA fragment loaders (CDNA5 ISA 7.12.2), each = 2x ds_load_b128.
// ---------------------------------------------------------------------------
static __device__ inline v16bf load_frag_a(const __bf16* __restrict__ s, int lane)
{
  const __bf16* r = s + (lane & 15) * LDT_S;
  const int kb = (lane & 16) ? 8 : 0;
  FragU f;
  f.q[0] = *reinterpret_cast<const uint4*>(r + kb);        // K kb..kb+7
  f.q[1] = *reinterpret_cast<const uint4*>(r + kb + 16);   // K kb+16..kb+23
  return f.v;
}

static __device__ inline v16bf load_frag_bt(const __bf16* __restrict__ s, int lane)
{
  const __bf16* r = s + (lane & 15) * LDT_S;
  const int kb = (lane & 16) ? 8 : 0;
  FragU f;
  f.q[0] = *reinterpret_cast<const uint4*>(r + kb);        // K kb..kb+7
  f.q[1] = *reinterpret_cast<const uint4*>(r + kb + 16);   // K kb+16..kb+23
  return f.v;
}

// ---------------------------------------------------------------------------
// bf16 WMMA GEMM, register + LDS double-buffered pipeline:
//   C[M,N] f32 = A[M,K] * Bt[N,K]^T, row-major, lda=ldb=K.
// Block tile 128x256, 512 threads = 16 waves (4M x 4N), wave tile 32x64
// (2x4 WMMA frags, 64 acc VGPRs) -> ~120 live VGPRs, no spills.
// mode 0: store f32 to Cout.  mode 1: Out = acc + D[n]*U[m,n].
// ---------------------------------------------------------------------------
__global__ __launch_bounds__(512, 1) void wmma_gemm(
    const __bf16* __restrict__ A, const __bf16* __restrict__ Bt,
    int M, int N, int K, float* __restrict__ Cout, int ldc, int mode,
    const float* __restrict__ Dvec, const float* __restrict__ U,
    float* __restrict__ Out)
{
  __shared__ __bf16 lsA[2][BM * LDT_S];   // 2 x 10240 B
  __shared__ __bf16 lsB[2][BN * LDT_S];   // 2 x 20480 B  (total 60 KB)

  const int tid  = threadIdx.x;
  const int lane = tid & 31;
  const int w    = tid >> 5;                    // 16 waves
  const int wm   = w & 3;                       // 4 M groups of 32 rows
  const int wn   = w >> 2;                      // 4 N groups of 64 cols
  const int m0   = blockIdx.y * BM;
  const int n0   = blockIdx.x * BN;

  const int srow = tid >> 2;                    // staging row (0..127)
  const int sseg = tid & 3;                     // 8-elem segment within 32-K row

  // persistent global staging pointers, advanced by BK each k-step
  const __bf16* gA;                             // A: 128 rows, 1 uint4/thread
  const __bf16* gB[2];                          // B: 256 rows, 2 uint4/thread
  gA = A + (size_t)(m0 + srow) * K + sseg * 8;
#pragma unroll
  for (int t = 0; t < 2; ++t)
    gB[t] = Bt + (size_t)(n0 + srow + t * 128) * K + sseg * 8;

  uint4 ra, rbq[2];                             // in-flight staged tile
  auto gload = [&]() {
    ra = *reinterpret_cast<const uint4*>(gA);  gA += BK;
#pragma unroll
    for (int t = 0; t < 2; ++t) { rbq[t] = *reinterpret_cast<const uint4*>(gB[t]); gB[t] += BK; }
  };
  auto sstore = [&](int buf) {
    *reinterpret_cast<uint4*>(&lsA[buf][srow * LDT_S + sseg * 8]) = ra;
#pragma unroll
    for (int t = 0; t < 2; ++t)
      *reinterpret_cast<uint4*>(&lsB[buf][(srow + t * 128) * LDT_S + sseg * 8]) = rbq[t];
  };

  v8f acc[2][4] = {};
  gload();

  const int T = K / BK;
  for (int i = 0; i < T; ++i) {
    sstore(i & 1);                              // waits only on loads from iter i-1
    __syncthreads();                            // publish tile i; drain frag reads
    if (i + 1 < T) gload();                     // next tile in flight during compute
    // speculative L2 hint one more tile ahead (ptr+imm, dropped if OOB)
    __builtin_prefetch(gA + BK, 0, 0);
    __builtin_prefetch(gB[0] + BK, 0, 0);

    const __bf16* bufA = lsA[i & 1];
    const __bf16* bufB = lsB[i & 1];

    v16bf af[2];
#pragma unroll
    for (int fm = 0; fm < 2; ++fm)
      af[fm] = load_frag_a(&bufA[(wm * 32 + fm * 16) * LDT_S], lane);
#pragma unroll
    for (int j = 0; j < 4; ++j) {
      v16bf bfr = load_frag_bt(&bufB[(wn * 64 + j * 16) * LDT_S], lane);
#pragma unroll
      for (int fm = 0; fm < 2; ++fm)
        acc[fm][j] = __builtin_amdgcn_wmma_f32_16x16x32_bf16(
            false, af[fm], false, bfr, (short)0, acc[fm][j], false, false);
    }
  }

  // epilogue: C/D layout -> lanes 0-15: M=r, lanes 16-31: M=8+r; N = lane%16
  const int col = lane & 15;
  const int rb  = (lane & 16) ? 8 : 0;
#pragma unroll
  for (int fm = 0; fm < 2; ++fm)
#pragma unroll
    for (int j = 0; j < 4; ++j) {
      int gn = n0 + wn * 64 + j * 16 + col;
#pragma unroll
      for (int r = 0; r < 8; ++r) {
        int gm = m0 + wm * 32 + fm * 16 + rb + r;
        float v = acc[fm][j][r];
        if (mode == 0) {
          Cout[(size_t)gm * ldc + gn] = v;
        } else {
          Out[(size_t)gm * ldc + gn] = v + Dvec[gn] * U[(size_t)gm * ldc + gn];
        }
      }
    }
}

// ---------------------------------------------------------------------------
// Chunked associative scan: x_l = Lbar_l * x_{l-1} + gamma_l * Bu_raw_l  (complex)
// ---------------------------------------------------------------------------
__global__ __launch_bounds__(256) void scan_pass1(
    const float* __restrict__ dts, const float* __restrict__ Lre,
    const float* __restrict__ Lim, const float* __restrict__ stepb,
    const float* __restrict__ Bu, float* __restrict__ agg)
{
  const int p  = threadIdx.x;
  const int ch = blockIdx.x;
  const int S  = NCH * P_DIM;
  const float lr = Lre[p], li = Lim[p], st = stepb[p];
  const float inv = 1.0f / (lr * lr + li * li);
  float Are = 1.f, Aim = 0.f, xre = 0.f, xim = 0.f;
  const int l0 = ch * CHUNK;
  for (int i = 0; i < CHUNK; ++i) {
    int l = l0 + i;
    float d  = dts[l] * st;
    float ea = __expf(lr * d);
    float sn, cs; __sincosf(li * d, &sn, &cs);
    float Ar = ea * cs, Ai = ea * sn;
    float t  = Ar - 1.0f;
    float gr = (t * lr + Ai * li) * inv;          // (Lbar-1)*conj(L)/|L|^2
    float gi = (Ai * lr - t * li) * inv;
    float br = Bu[(size_t)l * N2P + p];
    float bi = Bu[(size_t)l * N2P + P_DIM + p];
    float ur = gr * br - gi * bi;
    float ui = gr * bi + gi * br;
    float nr = Ar * xre - Ai * xim + ur;
    float ni = Ar * xim + Ai * xre + ui;
    xre = nr; xim = ni;
    float pr = Ar * Are - Ai * Aim;
    float pi = Ar * Aim + Ai * Are;
    Are = pr; Aim = pi;
  }
  agg[ch * P_DIM + p]         = Are;
  agg[S + ch * P_DIM + p]     = Aim;
  agg[2 * S + ch * P_DIM + p] = xre;
  agg[3 * S + ch * P_DIM + p] = xim;
}

__global__ __launch_bounds__(256) void scan_pass2(const float* __restrict__ agg,
                                                  float* __restrict__ pre)
{
  const int p = threadIdx.x;
  const int S = NCH * P_DIM;
  float sre = 0.f, sim = 0.f;
  for (int j = 0; j < NCH; ++j) {
    pre[j * P_DIM + p]     = sre;   // exclusive prefix (incoming state of chunk j)
    pre[S + j * P_DIM + p] = sim;
    float Ar = agg[j * P_DIM + p];
    float Ai = agg[S + j * P_DIM + p];
    float br = agg[2 * S + j * P_DIM + p];
    float bi = agg[3 * S + j * P_DIM + p];
    float nr = Ar * sre - Ai * sim + br;
    float ni = Ar * sim + Ai * sre + bi;
    sre = nr; sim = ni;
  }
}

__global__ __launch_bounds__(256) void scan_pass3(
    const float* __restrict__ dts, const float* __restrict__ Lre,
    const float* __restrict__ Lim, const float* __restrict__ stepb,
    const float* __restrict__ Bu, const float* __restrict__ pre,
    __bf16* __restrict__ X2)
{
  const int p  = threadIdx.x;
  const int ch = blockIdx.x;
  const int S  = NCH * P_DIM;
  const float lr = Lre[p], li = Lim[p], st = stepb[p];
  const float inv = 1.0f / (lr * lr + li * li);
  float xre = pre[ch * P_DIM + p];
  float xim = pre[S + ch * P_DIM + p];
  const int l0 = ch * CHUNK;
  for (int i = 0; i < CHUNK; ++i) {
    int l = l0 + i;
    float d  = dts[l] * st;
    float ea = __expf(lr * d);
    float sn, cs; __sincosf(li * d, &sn, &cs);
    float Ar = ea * cs, Ai = ea * sn;
    float t  = Ar - 1.0f;
    float gr = (t * lr + Ai * li) * inv;
    float gi = (Ai * lr - t * li) * inv;
    float br = Bu[(size_t)l * N2P + p];
    float bi = Bu[(size_t)l * N2P + P_DIM + p];
    float ur = gr * br - gi * bi;
    float ui = gr * bi + gi * br;
    float nr = Ar * xre - Ai * xim + ur;
    float ni = Ar * xim + Ai * xre + ui;
    xre = nr; xim = ni;
    X2[(size_t)l * N2P + p]         = f2bf(xre);   // sign/2x folded into W2t
    X2[(size_t)l * N2P + P_DIM + p] = f2bf(xim);
  }
}

// ---------------------------------------------------------------------------
// Launch
// ---------------------------------------------------------------------------
extern "C" void kernel_launch(void* const* d_in, const int* in_sizes, int n_in,
                              void* d_out, int out_size, void* d_ws, size_t ws_size,
                              hipStream_t stream)
{
  const float* u     = (const float*)d_in[0];
  const float* dts   = (const float*)d_in[1];
  const float* Lre   = (const float*)d_in[2];
  const float* Lim   = (const float*)d_in[3];
  const float* B_re  = (const float*)d_in[4];
  const float* B_im  = (const float*)d_in[5];
  const float* C_re  = (const float*)d_in[6];
  const float* C_im  = (const float*)d_in[7];
  const float* Dv    = (const float*)d_in[8];
  const float* lstep = (const float*)d_in[9];
  float* out = (float*)d_out;

  char* ws = (char*)d_ws;
  size_t off = 0;
  auto take = [&](size_t bytes) {
    size_t r = off;
    off = (off + bytes + 255) & ~(size_t)255;
    return r;
  };
  __bf16* ubf   = (__bf16*)(ws + take((size_t)L_SEQ * H_DIM * 2));
  __bf16* W1t   = (__bf16*)(ws + take((size_t)N2P * H_DIM * 2));
  __bf16* W2t   = (__bf16*)(ws + take((size_t)H_DIM * N2P * 2));
  float*  stepb = (float*) (ws + take((size_t)P_DIM * 4));
  float*  Bu    = (float*) (ws + take((size_t)L_SEQ * N2P * 4));
  __bf16* X2    = (__bf16*)(ws + take((size_t)L_SEQ * N2P * 2));
  float*  agg   = (float*) (ws + take((size_t)4 * NCH * P_DIM * 4));
  float*  pre   = (float*) (ws + take((size_t)2 * NCH * P_DIM * 4));

  prep_small<<<dim3((P_DIM * H_DIM) / 256), dim3(256), 0, stream>>>(
      lstep, B_re, B_im, C_re, C_im, stepb, W1t, W2t);
  cast_u<<<dim3((L_SEQ * H_DIM / 4) / 256), dim3(256), 0, stream>>>(u, ubf);

  // GEMM 1: Bu_raw[L,2P] = u[L,H] * W1t[2P,H]^T
  wmma_gemm<<<dim3(N2P / BN, L_SEQ / BM), dim3(512), 0, stream>>>(
      ubf, W1t, L_SEQ, N2P, H_DIM, Bu, N2P, 0, nullptr, nullptr, nullptr);

  scan_pass1<<<dim3(NCH), dim3(P_DIM), 0, stream>>>(dts, Lre, Lim, stepb, Bu, agg);
  scan_pass2<<<dim3(1), dim3(P_DIM), 0, stream>>>(agg, pre);
  scan_pass3<<<dim3(NCH), dim3(P_DIM), 0, stream>>>(dts, Lre, Lim, stepb, Bu, pre, X2);

  // GEMM 2: out[L,H] = X2[L,2P] * W2t[H,2P]^T + D*u
  wmma_gemm<<<dim3(H_DIM / BN, L_SEQ / BM), dim3(512), 0, stream>>>(
      X2, W2t, L_SEQ, H_DIM, N2P, nullptr, H_DIM, 1, Dv, u, out);
}
